// Transfromer_30743375905502
// MI455X (gfx1250) — matrix operand order repaired and checked by reading, not compiled, not run on hardware
//
#include <hip/hip_runtime.h>

// ---------------------------------------------------------------- types
typedef __attribute__((ext_vector_type(16))) __bf16 v16bf;
typedef __attribute__((ext_vector_type(8)))  float  v8f;
typedef __attribute__((ext_vector_type(4)))  int    v4i;

union FragU { v16bf v; uint4 q[2]; unsigned int u[8]; };
union FragC { v8f   v; float f[8]; };

#define S_  512
#define B_  32
#define D_  512
#define DH_ 64
#define FF_ 2048
#define L_  6
#define O_  1000
#define M_  (B_ * S_)          // 16384 rows of activations

// ---------------------------------------------------------------- async copy (CDNA5)
#if defined(__has_builtin)
#if __has_builtin(__builtin_amdgcn_global_load_async_to_lds_b128) && \
    __has_builtin(__builtin_amdgcn_s_wait_asynccnt)
#define HAVE_ASYNC_COPY 1
#endif
#endif
#ifndef HAVE_ASYNC_COPY
#define HAVE_ASYNC_COPY 0
#endif

#if HAVE_ASYNC_COPY
__device__ __forceinline__ void async_cp16(void* lds, const void* g) {
  __builtin_amdgcn_global_load_async_to_lds_b128(
      (__attribute__((address_space(1))) v4i*)g,
      (__attribute__((address_space(3))) v4i*)lds, 0, 0);
}
__device__ __forceinline__ void async_wait0() { __builtin_amdgcn_s_wait_asynccnt(0); }
#endif

// ---------------------------------------------------------------- helpers
__device__ __forceinline__ unsigned short f2bf(float f) {
  unsigned int u = __builtin_bit_cast(unsigned int, f);
  u += 0x7FFFu + ((u >> 16) & 1u);          // round-to-nearest-even
  return (unsigned short)(u >> 16);
}
__device__ __forceinline__ unsigned int packbf(float x, float y) {
  return (unsigned int)f2bf(x) | ((unsigned int)f2bf(y) << 16);
}

__device__ __forceinline__ v8f wmma_bf16(v16bf a, v16bf b, v8f c) {
  return __builtin_amdgcn_wmma_f32_16x16x32_bf16(false, a, false, b,
                                                 (short)0, c, false, false);
}

// A fragment (16x32 bf16): lane L<16 -> row L, K {0..7,16..23}; lane>=16 -> row L-16,
// K {8..15,24..31}. Row-major source viewed as uint4 (8 bf16 per quad).
__device__ __forceinline__ v16bf a_frag_q(const uint4* rowBaseQ, int strideQ,
                                          int lane, int kOffQ) {
  const uint4* p = rowBaseQ + (lane & 15) * strideQ + kOffQ;
  const int hl = lane >> 4;
  FragU f;
  f.q[0] = p[hl];
  f.q[1] = p[2 + hl];
  return f.v;
}
// B fragment (32x16 bf16): lane L -> column L%16, K {0..7,16..23} (L<16) / {8..15,24..31}.
__device__ __forceinline__ v16bf b_frag_q(const uint4* baseQ, int strideQ,
                                          int lane, int nOffQ) {
  const int se = strideQ * 8;
  const unsigned short* p = (const unsigned short*)baseQ + (lane >> 4) * 8 * se + nOffQ * 8 + (lane & 15);
  FragU f;
#pragma unroll
  for (int i = 0; i < 4; ++i) {
    f.u[i]     = (unsigned int)p[(2 * i) * se]      | ((unsigned int)p[(2 * i + 1) * se]      << 16);
    f.u[4 + i] = (unsigned int)p[(16 + 2 * i) * se] | ((unsigned int)p[(16 + 2 * i + 1) * se] << 16);
  }
  return f.v;
}
// ushort-element variants (used in flash for transposed K and the P round-trip)
__device__ __forceinline__ v16bf a_frag_u16(const unsigned short* s, int strideE,
                                            int lane, int kOff) {
  const unsigned short* p = s + (lane & 15) * strideE + kOff + (lane >> 4) * 8;
  FragU f;
#pragma unroll
  for (int i = 0; i < 4; ++i) {
    f.u[i]     = (unsigned int)p[2 * i]      | ((unsigned int)p[2 * i + 1]      << 16);
    f.u[4 + i] = (unsigned int)p[16 + 2 * i] | ((unsigned int)p[16 + 2 * i + 1] << 16);
  }
  return f.v;
}
__device__ __forceinline__ v16bf b_frag_u16(const unsigned short* s, int strideE,
                                            int lane, int nOff) {
  const unsigned short* p = s + (lane >> 4) * 8 * strideE + nOff + (lane & 15);
  FragU f;
#pragma unroll
  for (int i = 0; i < 4; ++i) {
    f.u[i]     = (unsigned int)p[(2 * i) * strideE]      | ((unsigned int)p[(2 * i + 1) * strideE]      << 16);
    f.u[4 + i] = (unsigned int)p[(16 + 2 * i) * strideE] | ((unsigned int)p[(16 + 2 * i + 1) * strideE] << 16);
  }
  return f.v;
}

// ---------------------------------------------------------------- small kernels
__global__ void f32_to_bf16_k(const float* __restrict__ src,
                              unsigned short* __restrict__ dst, int n2) {
  int i = blockIdx.x * 256 + threadIdx.x;
  if (i >= n2) return;
  ((unsigned int*)dst)[i] = packbf(src[2 * i], src[2 * i + 1]);
}

__global__ void embed_pe_k(const int* __restrict__ x, const float* __restrict__ emb,
                           float* __restrict__ actF, unsigned short* __restrict__ actB) {
  int row = blockIdx.x;               // row = b*S + s  (matches [B,S,D] layout)
  int b = row >> 9, s = row & 511;
  int tok = x[s * B_ + b];
#pragma unroll
  for (int part = 0; part < 2; ++part) {
    int d = threadIdx.x + part * 256;
    float ang = (float)s / powf(10000.0f, (float)((d >> 1) << 1) / (float)D_);
    float pe = (d & 1) ? cosf(ang) : sinf(ang);
    float v = emb[(size_t)tok * D_ + d] + pe;
    actF[(size_t)row * D_ + d] = v;
    actB[(size_t)row * D_ + d] = f2bf(v);
  }
}

__global__ void layernorm_k(const float* __restrict__ X, const float* __restrict__ g,
                            const float* __restrict__ b, float* __restrict__ Yf,
                            unsigned short* __restrict__ Yb) {
  __shared__ float red[256];
  int row = blockIdx.x, tid = threadIdx.x;
  const float* xr = X + (size_t)row * D_;
  float v0 = xr[tid], v1 = xr[tid + 256];
  red[tid] = v0 + v1;
  __syncthreads();
  for (int o = 128; o > 0; o >>= 1) { if (tid < o) red[tid] += red[tid + o]; __syncthreads(); }
  float mu = red[0] * (1.0f / D_);
  __syncthreads();
  float d0 = v0 - mu, d1 = v1 - mu;
  red[tid] = d0 * d0 + d1 * d1;
  __syncthreads();
  for (int o = 128; o > 0; o >>= 1) { if (tid < o) red[tid] += red[tid + o]; __syncthreads(); }
  float rs = rsqrtf(red[0] * (1.0f / D_) + 1e-5f);
  float o0 = d0 * rs * g[tid] + b[tid];
  float o1 = d1 * rs * g[tid + 256] + b[tid + 256];
  Yf[(size_t)row * D_ + tid] = o0;        Yb[(size_t)row * D_ + tid] = f2bf(o0);
  Yf[(size_t)row * D_ + tid + 256] = o1;  Yb[(size_t)row * D_ + tid + 256] = f2bf(o1);
}

// ---------------------------------------------------------------- GEMM (bf16 WMMA)
// C[M,N] = A[M,K] @ W[K,N]; block tile 64x64, K-step 64, 4 waves of 32x32.
// Double-buffered LDS: async copy of tile t+1 overlaps WMMAs on tile t.
// mode 0: outB = bf16(c+bias); 1: outB = bf16(relu(c+bias)); 2: outF = c+bias+resid.
__global__ __launch_bounds__(128)
void gemm_bf16_k(const unsigned short* __restrict__ A, const unsigned short* __restrict__ W,
                 const float* __restrict__ bias, const float* __restrict__ resid,
                 float* __restrict__ outF, unsigned short* __restrict__ outB,
                 int N, int K, int mode) {
  __shared__ uint4 a_s4[2][64 * 9];     // 64 rows x 64 bf16, row stride 9 quads (pad)
  __shared__ uint4 b_s4[2][64 * 9];     // 64 k-rows x 64 bf16
  const int KU4 = K >> 3, NU4 = N >> 3;
  const int m0 = blockIdx.y * 64, n0 = blockIdx.x * 64;
  const int tid = threadIdx.x, lane = tid & 31, w = tid >> 5;
  const int wm = (w >> 1) * 32, wn = (w & 1) * 32;
  const int sr = tid >> 3, sc = tid & 7;        // staging: 16 rows x 8 quads
  const uint4* pA = (const uint4*)A + (size_t)(m0 + sr) * KU4 + sc;
  const uint4* pB = (const uint4*)W + (size_t)sr * NU4 + (n0 >> 3) + sc;

  v8f z = {0.f, 0.f, 0.f, 0.f, 0.f, 0.f, 0.f, 0.f};
  v8f acc[2][2] = {{z, z}, {z, z}};
  const int nSteps = K >> 6;

  // ---- prologue: stage tile 0 into buffer 0
#if HAVE_ASYNC_COPY
#pragma unroll
  for (int i = 0; i < 4; ++i) {
    async_cp16(&a_s4[0][(sr + 16 * i) * 9 + sc], pA + (size_t)16 * i * KU4);
    async_cp16(&b_s4[0][(sr + 16 * i) * 9 + sc], pB + (size_t)16 * i * NU4);
  }
  async_wait0();
#else
  {
    uint4 ta[4], tb[4];
#pragma unroll
    for (int i = 0; i < 4; ++i) {
      ta[i] = pA[(size_t)16 * i * KU4];
      tb[i] = pB[(size_t)16 * i * NU4];
    }
#pragma unroll
    for (int i = 0; i < 4; ++i) {
      a_s4[0][(sr + 16 * i) * 9 + sc] = ta[i];
      b_s4[0][(sr + 16 * i) * 9 + sc] = tb[i];
    }
  }
#endif
  __syncthreads();
  pA += 8;
  pB += (size_t)64 * NU4;

  for (int t = 0; t < nSteps; ++t) {
    const int cur = t & 1, nxt = cur ^ 1;
    const bool more = (t + 1 < nSteps);
    // ---- stage tile t+1 into the alternate buffer (overlaps compute below)
#if HAVE_ASYNC_COPY
    if (more) {
#pragma unroll
      for (int i = 0; i < 4; ++i) {
        async_cp16(&a_s4[nxt][(sr + 16 * i) * 9 + sc], pA + (size_t)16 * i * KU4);
        async_cp16(&b_s4[nxt][(sr + 16 * i) * 9 + sc], pB + (size_t)16 * i * NU4);
      }
    }
#else
    uint4 ta[4], tb[4];
    if (more) {
#pragma unroll
      for (int i = 0; i < 4; ++i) {
        ta[i] = pA[(size_t)16 * i * KU4];
        tb[i] = pB[(size_t)16 * i * NU4];
      }
    }
#endif
    // ---- compute on tile t
#pragma unroll
    for (int c = 0; c < 2; ++c) {               // two 16x16x32 K-chunks
      v16bf af0 = a_frag_q(a_s4[cur] + wm * 9, 9, lane, c * 4);
      v16bf af1 = a_frag_q(a_s4[cur] + (wm + 16) * 9, 9, lane, c * 4);
      v16bf bf0 = b_frag_q(b_s4[cur] + (c * 32) * 9, 9, lane, wn >> 3);
      v16bf bf1 = b_frag_q(b_s4[cur] + (c * 32) * 9, 9, lane, (wn >> 3) + 2);
      acc[0][0] = wmma_bf16(af0, bf0, acc[0][0]);
      acc[0][1] = wmma_bf16(af0, bf1, acc[0][1]);
      acc[1][0] = wmma_bf16(af1, bf0, acc[1][0]);
      acc[1][1] = wmma_bf16(af1, bf1, acc[1][1]);
    }
#if HAVE_ASYNC_COPY
    if (more) async_wait0();
#else
    if (more) {
#pragma unroll
      for (int i = 0; i < 4; ++i) {
        a_s4[nxt][(sr + 16 * i) * 9 + sc] = ta[i];
        b_s4[nxt][(sr + 16 * i) * 9 + sc] = tb[i];
      }
    }
#endif
    __syncthreads();
    pA += 8;                                    // advance 64 bf16 along K
    pB += (size_t)64 * NU4;                     // advance 64 K-rows
  }

  const int hl = lane >> 4, r0 = lane & 15;
#pragma unroll
  for (int mt = 0; mt < 2; ++mt)
#pragma unroll
    for (int nt = 0; nt < 2; ++nt) {
      FragC c; c.v = acc[mt][nt];
#pragma unroll
      for (int r = 0; r < 8; ++r) {
        int gm = m0 + wm + mt * 16 + r + 8 * hl;
        int gn = n0 + wn + nt * 16 + r0;
        float v = c.f[r] + bias[gn];
        if (mode == 2) {
          outF[(size_t)gm * N + gn] = v + resid[(size_t)gm * N + gn];
        } else {
          if (mode == 1) v = fmaxf(v, 0.0f);
          outB[(size_t)gm * N + gn] = f2bf(v);
        }
      }
    }
}

// ---------------------------------------------------------------- flash attention
// q,k,v,ctx flat [256 groups][512 seq][64 dh] bf16 (reference's raw reshape).
// Block = 4 waves, 64 query rows (16/wave); KV streamed in 64-key tiles;
// K/V staging double-buffered (regs for K transpose, async LDS for V).
__global__ __launch_bounds__(128)
void flash_attn_k(const unsigned short* __restrict__ Q, const unsigned short* __restrict__ Kd,
                  const unsigned short* __restrict__ V, unsigned short* __restrict__ Ob,
                  float scale_) {
  __shared__ unsigned short kT_s[2][64 * 66]; // [d][key]  (B-frag for Q@K^T)
  __shared__ uint4          v_s4[2][64 * 9];  // [key][d]  (B-frag for P@V)
  __shared__ unsigned short p_s[4 * 16 * 66]; // per-wave P tile 16x64
  const int g = blockIdx.y, q0 = blockIdx.x * 64;
  const int tid = threadIdx.x, lane = tid & 31, w = tid >> 5;
  const int hl = lane >> 4, r0 = lane & 15;
  const int skey = tid >> 3, sdq = tid & 7;   // staging coords (x4 along key)
  const uint4* Q4 = (const uint4*)Q + (size_t)g * (S_ * DH_ / 8);
  const uint4* K4 = (const uint4*)Kd + (size_t)g * (S_ * DH_ / 8);
  const uint4* V4 = (const uint4*)V + (size_t)g * (S_ * DH_ / 8);

  v16bf aq[2];                                // Q fragment (K-dim 64 -> 2 chunks)
#pragma unroll
  for (int c = 0; c < 2; ++c)
    aq[c] = a_frag_q(Q4 + (size_t)(q0 + w * 16) * 8, 8, lane, c * 4);

  v8f z = {0.f, 0.f, 0.f, 0.f, 0.f, 0.f, 0.f, 0.f};
  v8f o[4] = {z, z, z, z};                    // 16 rows x 64 dh
  float m_[8], l_[8], alpha[8];
#pragma unroll
  for (int r = 0; r < 8; ++r) { m_[r] = -1e30f; l_[r] = 0.0f; }

  // ---- prologue: stage KV tile 0 into buffer 0
  {
    uint4 kq[4];
#pragma unroll
    for (int i = 0; i < 4; ++i) {
      int key = skey + 16 * i;
      kq[i] = K4[(size_t)key * 8 + sdq];
#if HAVE_ASYNC_COPY
      async_cp16(&v_s4[0][key * 9 + sdq], V4 + (size_t)key * 8 + sdq);
#endif
    }
#if !HAVE_ASYNC_COPY
    uint4 vq[4];
#pragma unroll
    for (int i = 0; i < 4; ++i) vq[i] = V4[(size_t)(skey + 16 * i) * 8 + sdq];
#endif
#pragma unroll
    for (int i = 0; i < 4; ++i) {
      int key = skey + 16 * i, d0 = 8 * sdq;
      unsigned int cc[4] = {kq[i].x, kq[i].y, kq[i].z, kq[i].w};
#pragma unroll
      for (int c2 = 0; c2 < 4; ++c2) {
        kT_s[0][(d0 + 2 * c2) * 66 + key]     = (unsigned short)(cc[c2] & 0xFFFFu);
        kT_s[0][(d0 + 2 * c2 + 1) * 66 + key] = (unsigned short)(cc[c2] >> 16);
      }
    }
#if HAVE_ASYNC_COPY
    async_wait0();
#else
#pragma unroll
    for (int i = 0; i < 4; ++i) v_s4[0][(skey + 16 * i) * 9 + sdq] = vq[i];
#endif
  }
  __syncthreads();

  for (int kt = 0; kt < 8; ++kt) {
    const int cur = kt & 1, nxt = cur ^ 1;
    const bool more = (kt < 7);
    const int j1 = (kt + 1) * 64;
    // ---- prefetch next KV tile (overlaps compute below)
    uint4 kqn[4];
#if !HAVE_ASYNC_COPY
    uint4 vqn[4];
#endif
    if (more) {
#pragma unroll
      for (int i = 0; i < 4; ++i) {
        int key = skey + 16 * i;
        kqn[i] = K4[(size_t)(j1 + key) * 8 + sdq];
#if HAVE_ASYNC_COPY
        async_cp16(&v_s4[nxt][key * 9 + sdq], V4 + (size_t)(j1 + key) * 8 + sdq);
#else
        vqn[i] = V4[(size_t)(j1 + key) * 8 + sdq];
#endif
      }
    }

    // ---- scores: 16 q x 64 keys from kT_s[cur]
    v8f sc[4] = {z, z, z, z};
#pragma unroll
    for (int nt = 0; nt < 4; ++nt)
#pragma unroll
      for (int c = 0; c < 2; ++c) {
        v16bf bk = b_frag_u16(kT_s[cur] + (c * 32) * 66, 66, lane, nt * 16);
        sc[nt] = wmma_bf16(aq[c], bk, sc[nt]);
      }

    // ---- online softmax
    FragC scf[4];
#pragma unroll
    for (int nt = 0; nt < 4; ++nt) {
      scf[nt].v = sc[nt];
#pragma unroll
      for (int r = 0; r < 8; ++r) scf[nt].f[r] *= scale_;
    }
#pragma unroll
    for (int r = 0; r < 8; ++r) {
      float rm = fmaxf(fmaxf(scf[0].f[r], scf[1].f[r]), fmaxf(scf[2].f[r], scf[3].f[r]));
#pragma unroll
      for (int off = 1; off < 16; off <<= 1) rm = fmaxf(rm, __shfl_xor(rm, off, 16));
      float mn = fmaxf(m_[r], rm);
      alpha[r] = __expf(m_[r] - mn);
      m_[r] = mn;
    }
    float rsum[8];
#pragma unroll
    for (int r = 0; r < 8; ++r) rsum[r] = 0.0f;
#pragma unroll
    for (int nt = 0; nt < 4; ++nt)
#pragma unroll
      for (int r = 0; r < 8; ++r) {
        float pv = __expf(scf[nt].f[r] - m_[r]);
        scf[nt].f[r] = pv;
        rsum[r] += pv;
      }
#pragma unroll
    for (int r = 0; r < 8; ++r) {
#pragma unroll
      for (int off = 1; off < 16; off <<= 1) rsum[r] += __shfl_xor(rsum[r], off, 16);
      l_[r] = l_[r] * alpha[r] + rsum[r];
    }
#pragma unroll
    for (int dt = 0; dt < 4; ++dt) {
      FragC oc; oc.v = o[dt];
#pragma unroll
      for (int r = 0; r < 8; ++r) oc.f[r] *= alpha[r];
      o[dt] = oc.v;
    }
    // ---- P (C-layout) -> A-layout via per-wave LDS round trip
    unsigned short* pw = p_s + w * (16 * 66);
#pragma unroll
    for (int nt = 0; nt < 4; ++nt)
#pragma unroll
      for (int r = 0; r < 8; ++r)
        pw[(r + 8 * hl) * 66 + nt * 16 + r0] = f2bf(scf[nt].f[r]);
    __syncthreads();
#pragma unroll
    for (int c = 0; c < 2; ++c) {
      v16bf ap = a_frag_u16(pw, 66, lane, c * 32);
#pragma unroll
      for (int dt = 0; dt < 4; ++dt) {
        v16bf bv = b_frag_q(v_s4[cur] + (c * 32) * 9, 9, lane, 2 * dt);
        o[dt] = wmma_bf16(ap, bv, o[dt]);
      }
    }
    // ---- commit next tile's K transpose (+ V in fallback), then sync
    if (more) {
#pragma unroll
      for (int i = 0; i < 4; ++i) {
        int key = skey + 16 * i, d0 = 8 * sdq;
        unsigned int cc[4] = {kqn[i].x, kqn[i].y, kqn[i].z, kqn[i].w};
#pragma unroll
        for (int c2 = 0; c2 < 4; ++c2) {
          kT_s[nxt][(d0 + 2 * c2) * 66 + key]     = (unsigned short)(cc[c2] & 0xFFFFu);
          kT_s[nxt][(d0 + 2 * c2 + 1) * 66 + key] = (unsigned short)(cc[c2] >> 16);
        }
      }
#if !HAVE_ASYNC_COPY
#pragma unroll
      for (int i = 0; i < 4; ++i) v_s4[nxt][(skey + 16 * i) * 9 + sdq] = vqn[i];
#endif
    }
#if HAVE_ASYNC_COPY
    if (more) async_wait0();
#endif
    __syncthreads();
  }
  float inv[8];
#pragma unroll
  for (int r = 0; r < 8; ++r) inv[r] = 1.0f / l_[r];
#pragma unroll
  for (int dt = 0; dt < 4; ++dt) {
    FragC oc; oc.v = o[dt];
#pragma unroll
    for (int r = 0; r < 8; ++r) {
      int row = q0 + w * 16 + r + 8 * hl;
      Ob[(size_t)g * (S_ * DH_) + (size_t)row * DH_ + dt * 16 + r0] = f2bf(oc.f[r] * inv[r]);
    }
  }
}

// ---------------------------------------------------------------- final FC (split-K WMMA)
// A = act_bf [32, 262144] bf16, W_fc fp32 [262144,1000] converted in-register.
__global__ __launch_bounds__(128)
void fc_splitk_k(const unsigned short* __restrict__ A, const float* __restrict__ Wfc,
                 float* __restrict__ part) {
  const int nt = blockIdx.x, c = blockIdx.y;
  const int tid = threadIdx.x, lane = tid & 31, w = tid >> 5;
  const int hl = lane >> 4, r0 = lane & 15;
  const int n0 = nt * 16;
  const int kbase = c * 4096 + w * 1024;
  const uint4* A4 = (const uint4*)A;           // row stride 32768 quads
  const int AK4 = S_ * D_ / 8;
  v8f z = {0.f, 0.f, 0.f, 0.f, 0.f, 0.f, 0.f, 0.f};
  v8f acc[2] = {z, z};
  const bool full = (n0 + 16 <= O_);
#pragma unroll 2
  for (int ks = 0; ks < 1024; ks += 32) {
    const int k0 = kbase + ks;
    const int krow = k0 + hl * 8;              // B fragment: lane = column, K {8h..,16+8h..}
    const float* wp = Wfc + (size_t)krow * O_ + n0 + r0;
    if (ks + 32 < 1024)
      __builtin_prefetch(wp + (size_t)32 * O_, 0, 1);
    FragU fb;
    if (full) {
#pragma unroll
      for (int i = 0; i < 4; ++i) {
        fb.u[i]     = packbf(wp[(size_t)(2 * i) * O_], wp[(size_t)(2 * i + 1) * O_]);
        fb.u[4 + i] = packbf(wp[(size_t)(16 + 2 * i) * O_], wp[(size_t)(17 + 2 * i) * O_]);
      }
    } else {
      const bool ok = (n0 + r0 < O_);
#pragma unroll
      for (int i = 0; i < 4; ++i) {
        fb.u[i]     = ok ? packbf(wp[(size_t)(2 * i) * O_], wp[(size_t)(2 * i + 1) * O_]) : 0u;
        fb.u[4 + i] = ok ? packbf(wp[(size_t)(16 + 2 * i) * O_], wp[(size_t)(17 + 2 * i) * O_]) : 0u;
      }
    }
    v16bf bfr = fb.v;
#pragma unroll
    for (int mt = 0; mt < 2; ++mt) {
      v16bf af = a_frag_q(A4 + (size_t)mt * 16 * AK4, AK4, lane, k0 >> 3);
      acc[mt] = wmma_bf16(af, bfr, acc[mt]);
    }
  }
  const int slice = c * 4 + w;
  float* pd = part + ((size_t)(slice * 63 + nt) * 32) * 16;
#pragma unroll
  for (int mt = 0; mt < 2; ++mt) {
    FragC cc; cc.v = acc[mt];
#pragma unroll
    for (int r = 0; r < 8; ++r)
      pd[(mt * 16 + r + 8 * hl) * 16 + r0] = cc.f[r];
  }
}

__global__ void fc_reduce_k(const float* __restrict__ part, const float* __restrict__ bfc,
                            float* __restrict__ out) {
  int i = blockIdx.x * 256 + threadIdx.x;      // 32 x 1008 lattice
  int m = i / 1008, n = i % 1008;
  if (m >= 32 || n >= O_) return;
  int ntile = n >> 4, nc = n & 15;
  float s = bfc[n];
  for (int sl = 0; sl < 256; ++sl)
    s += part[((size_t)(sl * 63 + ntile) * 32 + m) * 16 + nc];
  out[(size_t)m * O_ + n] = s;
}

// ---------------------------------------------------------------- launch
extern "C" void kernel_launch(void* const* d_in, const int* in_sizes, int n_in,
                              void* d_out, int out_size, void* d_ws, size_t ws_size,
                              hipStream_t stream) {
  (void)in_sizes; (void)n_in; (void)out_size; (void)ws_size;
  const int*   x    = (const int*)d_in[0];
  const float* emb  = (const float*)d_in[2];
  const float* Wq   = (const float*)d_in[3];
  const float* bq   = (const float*)d_in[4];
  const float* Wk   = (const float*)d_in[5];
  const float* bk   = (const float*)d_in[6];
  const float* Wv   = (const float*)d_in[7];
  const float* bv   = (const float*)d_in[8];
  const float* Wo   = (const float*)d_in[9];
  const float* bo   = (const float*)d_in[10];
  const float* ln1g = (const float*)d_in[11];
  const float* ln1b = (const float*)d_in[12];
  const float* W1   = (const float*)d_in[13];
  const float* b1   = (const float*)d_in[14];
  const float* W2   = (const float*)d_in[15];
  const float* b2   = (const float*)d_in[16];
  const float* ln2g = (const float*)d_in[17];
  const float* ln2b = (const float*)d_in[18];
  const float* Wfc  = (const float*)d_in[19];
  const float* bfc  = (const float*)d_in[20];
  float* outp = (float*)d_out;

  char* p = (char*)d_ws;
  float* actA = (float*)p;                      p += (size_t)M_ * D_ * 4;
  float* actB = (float*)p;                      p += (size_t)M_ * D_ * 4;
  unsigned short* act_bf = (unsigned short*)p;  p += (size_t)M_ * D_ * 2;
  unsigned short* q_bf   = (unsigned short*)p;  p += (size_t)M_ * D_ * 2;
  unsigned short* k_bf   = (unsigned short*)p;  p += (size_t)M_ * D_ * 2;
  unsigned short* v_bf   = (unsigned short*)p;  p += (size_t)M_ * D_ * 2;
  unsigned short* ctx_bf = (unsigned short*)p;  p += (size_t)M_ * D_ * 2;
  unsigned short* ff_bf  = (unsigned short*)p;  p += (size_t)M_ * FF_ * 2;
  unsigned short* wqb    = (unsigned short*)p;  p += (size_t)L_ * D_ * D_ * 2;
  unsigned short* wkb    = (unsigned short*)p;  p += (size_t)L_ * D_ * D_ * 2;
  unsigned short* wvb    = (unsigned short*)p;  p += (size_t)L_ * D_ * D_ * 2;
  unsigned short* wob    = (unsigned short*)p;  p += (size_t)L_ * D_ * D_ * 2;
  unsigned short* w1b    = (unsigned short*)p;  p += (size_t)L_ * D_ * FF_ * 2;
  unsigned short* w2b    = (unsigned short*)p;  p += (size_t)L_ * FF_ * D_ * 2;
  float* partb = (float*)p;                     p += (size_t)256 * 63 * 32 * 16 * 4;

  const int nDD2  = L_ * D_ * D_ / 2;
  const int nDFF2 = L_ * D_ * FF_ / 2;
  f32_to_bf16_k<<<(nDD2 + 255) / 256, 256, 0, stream>>>(Wq, wqb, nDD2);
  f32_to_bf16_k<<<(nDD2 + 255) / 256, 256, 0, stream>>>(Wk, wkb, nDD2);
  f32_to_bf16_k<<<(nDD2 + 255) / 256, 256, 0, stream>>>(Wv, wvb, nDD2);
  f32_to_bf16_k<<<(nDD2 + 255) / 256, 256, 0, stream>>>(Wo, wob, nDD2);
  f32_to_bf16_k<<<(nDFF2 + 255) / 256, 256, 0, stream>>>(W1, w1b, nDFF2);
  f32_to_bf16_k<<<(nDFF2 + 255) / 256, 256, 0, stream>>>(W2, w2b, nDFF2);

  embed_pe_k<<<M_, 256, 0, stream>>>(x, emb, actA, act_bf);

  dim3 g512(D_ / 64, M_ / 64), g2048(FF_ / 64, M_ / 64);
  for (int l = 0; l < L_; ++l) {
    const unsigned short* wq_l = wqb + (size_t)l * D_ * D_;
    const unsigned short* wk_l = wkb + (size_t)l * D_ * D_;
    const unsigned short* wv_l = wvb + (size_t)l * D_ * D_;
    const unsigned short* wo_l = wob + (size_t)l * D_ * D_;
    const unsigned short* w1_l = w1b + (size_t)l * D_ * FF_;
    const unsigned short* w2_l = w2b + (size_t)l * FF_ * D_;
    gemm_bf16_k<<<g512, 128, 0, stream>>>(act_bf, wq_l, bq + l * D_, nullptr, nullptr, q_bf, D_, D_, 0);
    gemm_bf16_k<<<g512, 128, 0, stream>>>(act_bf, wk_l, bk + l * D_, nullptr, nullptr, k_bf, D_, D_, 0);
    gemm_bf16_k<<<g512, 128, 0, stream>>>(act_bf, wv_l, bv + l * D_, nullptr, nullptr, v_bf, D_, D_, 0);
    flash_attn_k<<<dim3(S_ / 64, B_ * 8), 128, 0, stream>>>(q_bf, k_bf, v_bf, ctx_bf, 0.125f);
    gemm_bf16_k<<<g512, 128, 0, stream>>>(ctx_bf, wo_l, bo + l * D_, actA, actB, nullptr, D_, D_, 2);
    layernorm_k<<<M_, 256, 0, stream>>>(actB, ln1g + l * D_, ln1b + l * D_, actA, act_bf);
    gemm_bf16_k<<<g2048, 128, 0, stream>>>(act_bf, w1_l, b1 + l * FF_, nullptr, nullptr, ff_bf, FF_, D_, 1);
    gemm_bf16_k<<<g512, 128, 0, stream>>>(ff_bf, w2_l, b2 + l * D_, actA, actB, nullptr, D_, FF_, 2);
    layernorm_k<<<M_, 256, 0, stream>>>(actB, ln2g + l * D_, ln2b + l * D_, actA, act_bf);
  }

  fc_splitk_k<<<dim3(63, 64), 128, 0, stream>>>(act_bf, Wfc, partb);
  fc_reduce_k<<<(32 * 1008 + 255) / 256, 256, 0, stream>>>(partb, bfc, outp);
}
